// S4Layer_66864050864258
// MI455X (gfx1250) — matrix operand order repaired and checked by reading, not compiled, not run on hardware
//
#include <hip/hip_runtime.h>

typedef __attribute__((ext_vector_type(16))) _Float16 v16h;
typedef __attribute__((ext_vector_type(8)))  _Float16 v8h;
typedef __attribute__((ext_vector_type(8)))  float    v8f;

#define L_SEQ   4096
#define D_MODEL 512
#define NST     64
#define CHUNK   64
#define NCHUNK  (L_SEQ / CHUNK)

// workspace layout (halves): four 64x64 f16 row-major matrices (32 KB total)
#define K_OFF 0
#define W_OFF 4096
#define P_OFF 8192
#define M_OFF 12288

// ---------------------------------------------------------------------------
// Kernel 1: discretization + chunk-operator precompute (runs once, 1 block)
// ---------------------------------------------------------------------------
__global__ __launch_bounds__(256) void s4_precompute(
    const float* __restrict__ A, const float* __restrict__ Bin,
    const float* __restrict__ Cin, _Float16* __restrict__ ws)
{
    __shared__ float aug[64 * 132];   // [E | F | b], row stride 132
    __shared__ float sPb[64 * 64];
    __shared__ float mult[64];
    __shared__ float rcur[64], rnext[64], ccur[64], cnext[64], sBv[64], kv[64];
    __shared__ float invp;

    const int tid = threadIdx.x;
    const float step = 1.0f / (float)L_SEQ;

    // Build augmented system  [I - s/2 A | I + s/2 A | s*B]
    for (int idx = tid; idx < 64 * 129; idx += 256) {
        int r = idx / 129, c = idx % 129;
        float v;
        if (c < 64)        v = ((r == c) ? 1.0f : 0.0f) - 0.5f * step * A[r * 64 + c];
        else if (c < 128) { int c2 = c - 64;
                           v = ((r == c2) ? 1.0f : 0.0f) + 0.5f * step * A[r * 64 + c2]; }
        else               v = step * Bin[r];
        aug[r * 132 + c] = v;
    }
    __syncthreads();

    // Gauss-Jordan (E is diagonally dominant ~ I, no pivoting needed)
    for (int p = 0; p < 64; ++p) {
        if (tid == 0) invp = 1.0f / aug[p * 132 + p];
        __syncthreads();
        for (int c = tid; c < 129; c += 256) aug[p * 132 + c] *= invp;
        if (tid < 64) mult[tid] = (tid == p) ? 0.0f : aug[tid * 132 + p];
        __syncthreads();
        for (int idx = tid; idx < 64 * 129; idx += 256) {
            int r = idx / 129, c = idx % 129;
            if (r != p) aug[r * 132 + c] -= mult[r] * aug[p * 132 + c];
        }
        __syncthreads();
    }
    // Now: Ab[i][j] = aug[i*132 + 64 + j], Bv[i] = aug[i*132 + 128]

    if (tid < 64) {
        sBv[tid]  = aug[tid * 132 + 128];
        rcur[tid] = Cin[tid];      // r_0 = Cv
        ccur[tid] = sBv[tid];      // c_0 = Bv
    }
    __syncthreads();

    // Iterate r_{m+1} = r_m Ab,  c_{m+1} = Ab c_m; emit k_m, M, W
    for (int m = 0; m < 64; ++m) {
        if (tid == 0) {
            float a = 0.0f;
            for (int i = 0; i < 64; ++i) a += rcur[i] * sBv[i];
            kv[m] = a;                                   // k_m = Cv Ab^m Bv
        }
        if (tid < 64) {
            ws[M_OFF + tid * 64 + (63 - m)] = (_Float16)ccur[tid];  // M[:,63-m]=c_m
            float ra = 0.0f, ca = 0.0f;
            for (int i = 0; i < 64; ++i) {
                ra += rcur[i] * aug[i * 132 + 64 + tid];
                ca += aug[tid * 132 + 64 + i] * ccur[i];
            }
            rnext[tid] = ra; cnext[tid] = ca;
        }
        __syncthreads();
        if (tid < 64) {
            ws[W_OFF + m * 64 + tid] = (_Float16)rnext[tid];        // W[m] = r_{m+1}
            rcur[tid] = rnext[tid]; ccur[tid] = cnext[tid];
        }
        __syncthreads();
    }

    // Kc[t][j] = k_{t-j} for j<=t else 0
    for (int idx = tid; idx < 4096; idx += 256) {
        int t = idx >> 6, j = idx & 63;
        ws[K_OFF + idx] = (_Float16)((j <= t) ? kv[t - j] : 0.0f);
    }

    // P = Ab^64 via 6 squarings; Q0 reuses aug cols 0..63 (stride 132)
    for (int idx = tid; idx < 4096; idx += 256) {
        int i = idx >> 6, j = idx & 63;
        aug[i * 132 + j] = aug[i * 132 + 64 + j];
    }
    __syncthreads();
    for (int itn = 0; itn < 6; ++itn) {
        bool fromQ0 = ((itn & 1) == 0);
        for (int idx = tid; idx < 4096; idx += 256) {
            int i = idx >> 6, j = idx & 63;
            float acc = 0.0f;
            if (fromQ0) {
                for (int k = 0; k < 64; ++k) acc += aug[i * 132 + k] * aug[k * 132 + j];
                sPb[i * 64 + j] = acc;
            } else {
                for (int k = 0; k < 64; ++k) acc += sPb[i * 64 + k] * sPb[k * 64 + j];
                aug[i * 132 + j] = acc;
            }
        }
        __syncthreads();
    }
    // after iteration 5 the result lives in Q0
    for (int idx = tid; idx < 4096; idx += 256)
        ws[P_OFF + idx] = (_Float16)aug[(idx >> 6) * 132 + (idx & 63)];
}

// ---------------------------------------------------------------------------
// Kernel 2: chunked WMMA scan. 256 blocks x 128 threads (4 waves).
// Block owns 16 sequences (same b, consecutive d). Wave w owns output-row
// tile [16w,16w+16) of both the Y GEMM (time rows) and the X GEMM (state rows).
// ---------------------------------------------------------------------------
__device__ __forceinline__ v16h frag_a(const _Float16* m, int row, int k0, int lane)
{
    // A fragment in contiguous k order (same as frag_b): lanes 0-15 hold K=0..15,
    // lanes 16-31 hold K=16..31 of row M=lane&15.
    const int base = (lane & 16) ? 16 : 0;
    return *(const v16h*)(m + row * 64 + k0 + base);
}

__device__ __forceinline__ v16h frag_b(const _Float16* m, int k0, int lane)
{
    // B-matrix 32x16: lanes 0-15 = col N=lane, K=k0..k0+15;
    // lanes 16-31 = col N=lane-16, K=k0+16..k0+31. m is stored transposed
    // ([N][K], K contiguous) so this is one 32-byte contiguous read.
    const int n = lane & 15;
    const int kk = k0 + ((lane & 16) ? 16 : 0);
    return *(const v16h*)(m + n * 64 + kk);
}

__global__ __launch_bounds__(128) void s4_scan(
    const float* __restrict__ u, const _Float16* __restrict__ ws,
    float* __restrict__ y)
{
    __shared__ __align__(32) _Float16 suT[16 * 64];   // u chunk, [seq][t]
    __shared__ __align__(32) _Float16 sxT[16 * 64];   // state,   [seq][n]

    const int tid  = threadIdx.x;
    const int lane = tid & 31;
    const int wave = tid >> 5;
    const int bIdx = blockIdx.x >> 5;          // 32 blocks per batch element
    const int d0   = (blockIdx.x & 31) << 4;   // 16 consecutive d per block
    const int rowBase = wave << 4;
    const int fr = lane & 15;

    // Chunk-invariant A-operand fragments (loaded once from global f16)
    const v16h fK0 = frag_a(ws + K_OFF, rowBase + fr, 0,  lane);
    const v16h fK1 = frag_a(ws + K_OFF, rowBase + fr, 32, lane);
    const v16h fW0 = frag_a(ws + W_OFF, rowBase + fr, 0,  lane);
    const v16h fW1 = frag_a(ws + W_OFF, rowBase + fr, 32, lane);
    const v16h fP0 = frag_a(ws + P_OFF, rowBase + fr, 0,  lane);
    const v16h fP1 = frag_a(ws + P_OFF, rowBase + fr, 32, lane);
    const v16h fM0 = frag_a(ws + M_OFF, rowBase + fr, 0,  lane);
    const v16h fM1 = frag_a(ws + M_OFF, rowBase + fr, 32, lane);

    // x0 = 0; preload u chunk 0 (transposed + f16 convert)
    for (int i = tid; i < 16 * 64; i += 128) sxT[i] = (_Float16)0.0f;
    {
        const int s  = tid & 15;
        const int t0 = (tid >> 4) << 3;
        const float* up = u + ((size_t)bIdx * L_SEQ + t0) * D_MODEL + d0 + s;
        #pragma unroll
        for (int j = 0; j < 8; ++j)
            suT[s * 64 + t0 + j] = (_Float16)up[(size_t)j * D_MODEL];
    }
    __syncthreads();

    for (int ch = 0; ch < NCHUNK; ++ch) {
        const v16h fu0 = frag_b(suT, 0,  lane);
        const v16h fu1 = frag_b(suT, 32, lane);
        const v16h fx0 = frag_b(sxT, 0,  lane);
        const v16h fx1 = frag_b(sxT, 32, lane);

        // Y = Kc*U + W*X0
        v8f yacc = {};
        yacc = __builtin_amdgcn_wmma_f32_16x16x32_f16(false, fK0, false, fu0, (short)0, yacc, false, false);
        yacc = __builtin_amdgcn_wmma_f32_16x16x32_f16(false, fK1, false, fu1, (short)0, yacc, false, false);
        yacc = __builtin_amdgcn_wmma_f32_16x16x32_f16(false, fW0, false, fx0, (short)0, yacc, false, false);
        yacc = __builtin_amdgcn_wmma_f32_16x16x32_f16(false, fW1, false, fx1, (short)0, yacc, false, false);

        // Xnext = P*X0 + M*U
        v8f xacc = {};
        xacc = __builtin_amdgcn_wmma_f32_16x16x32_f16(false, fP0, false, fx0, (short)0, xacc, false, false);
        xacc = __builtin_amdgcn_wmma_f32_16x16x32_f16(false, fP1, false, fx1, (short)0, xacc, false, false);
        xacc = __builtin_amdgcn_wmma_f32_16x16x32_f16(false, fM0, false, fu0, (short)0, xacc, false, false);
        xacc = __builtin_amdgcn_wmma_f32_16x16x32_f16(false, fM1, false, fu1, (short)0, xacc, false, false);

        __syncthreads();   // all waves finished reading suT/sxT

        // commit new state: D layout -> [seq][n] in LDS
        {
            const int nB = rowBase + ((lane & 16) ? 8 : 0);
            #pragma unroll
            for (int r = 0; r < 8; ++r)
                sxT[fr * 64 + nB + r] = (_Float16)xacc[r];
        }
        // prefetch next u chunk
        if (ch + 1 < NCHUNK) {
            const int s  = tid & 15;
            const int t0 = (tid >> 4) << 3;
            const float* up = u + ((size_t)bIdx * L_SEQ + (ch + 1) * CHUNK + t0) * D_MODEL + d0 + s;
            #pragma unroll
            for (int j = 0; j < 8; ++j)
                suT[s * 64 + t0 + j] = (_Float16)up[(size_t)j * D_MODEL];
        }
        // store Y tile: lanes of fixed time row cover 16 consecutive d
        {
            const int tB = ch * CHUNK + rowBase + ((lane & 16) ? 8 : 0);
            float* yp = y + ((size_t)bIdx * L_SEQ + tB) * D_MODEL + d0 + fr;
            #pragma unroll
            for (int r = 0; r < 8; ++r)
                yp[(size_t)r * D_MODEL] = yacc[r];
        }
        __syncthreads();
    }
}

// ---------------------------------------------------------------------------
extern "C" void kernel_launch(void* const* d_in, const int* in_sizes, int n_in,
                              void* d_out, int out_size, void* d_ws, size_t ws_size,
                              hipStream_t stream)
{
    (void)in_sizes; (void)n_in; (void)out_size; (void)ws_size;
    const float* u = (const float*)d_in[0];
    const float* A = (const float*)d_in[1];
    const float* B = (const float*)d_in[2];
    const float* C = (const float*)d_in[3];
    float* yout    = (float*)d_out;
    _Float16* ws   = (_Float16*)d_ws;   // needs 16384 halves = 32 KB

    s4_precompute<<<dim3(1),   dim3(256), 0, stream>>>(A, B, C, ws);
    s4_scan      <<<dim3(256), dim3(128), 0, stream>>>(u, ws, yout);
}
